// MixtureModel_5789615915587
// MI455X (gfx1250) — compile-verified, matrix operand order checked by reading, not run on hardware
//
#include <hip/hip_runtime.h>
#include <hip/hip_bf16.h>

typedef __attribute__((ext_vector_type(16))) _Float16 v16h;
typedef __attribute__((ext_vector_type(8)))  float    v8f;
typedef __attribute__((ext_vector_type(4)))  unsigned int u32x4;
typedef __attribute__((ext_vector_type(8)))  int      i32x8;
typedef __attribute__((ext_vector_type(4)))  int      i32x4;
typedef unsigned int u32;

// Problem constants (match reference)
constexpr int KC = 8;       // mixture components
constexpr int D  = 512;     // obs dim
constexpr int H  = 512;     // hidden dim
constexpr int AC = 1024;    // action dim
constexpr int SQ = 64;      // sequences
constexpr int TT = 128;     // timesteps
constexpr int BB = SQ * TT; // 8192 rows

// ---------------------------------------------------------------------------
// WMMA helpers (gfx1250, wave32, 16x16x32 f16 -> f32)
// ---------------------------------------------------------------------------
__device__ __forceinline__ v8f wmma16(v16h a, v16h b, v8f c) {
  // (neg_a, A, neg_b, B, c_mod, C, reuse_a, reuse_b)
  return __builtin_amdgcn_wmma_f32_16x16x32_f16(false, a, false, b, (short)0, c,
                                                false, false);
}

// K offset inside a 16x32 A-frag for VGPR v and lane half (ISA 7.12.2 table)
__device__ __forceinline__ int koff_of(int v, int laneHi8) {
  return (v < 4 ? 2 * v : 16 + 2 * (v - 4)) + laneHi8;
}

// Inverse mapping: element (M, c) -> half offset inside packed A-frag storage
// (one frag = 512 halfs = 1KB, lane-contiguous 32B per lane).
__device__ __forceinline__ int apack_off(int M, int c) {
  const int kt = c >> 5, cw = c & 31;
  const int half = (cw >> 3) & 1;
  const int v = ((cw >> 4) << 2) + ((cw & 7) >> 1);
  return kt * 512 + (M + (half << 4)) * 16 + (v << 1) + (cw & 1);
}

// Load packed A-fragment (lane's 16 halfs contiguous -> 2x ds_load_b128).
__device__ __forceinline__ v16h load_a_packed(const _Float16* pk, int kt,
                                              int lane) {
  return *(const v16h*)(pk + (size_t)kt * 512 + lane * 16);
}

// Load 32x16 f16 B-fragment from pre-packed layout (2x global_load_b128).
__device__ __forceinline__ v16h load_b(const _Float16* packed, int numKT,
                                       int nt, int kt, int lane) {
  const _Float16* p = packed + ((((size_t)nt * numKT + kt) * 32 + lane) * 16);
  return *(const v16h*)p;
}

// ---------------------------------------------------------------------------
// TDM: stage a contiguous block (global -> LDS offset 0) via the Tensor Data
// Mover.  D# per ISA 08_async_tensor 8.3/8.4: 1 x nbytes tile, data_size=8B.
// Caller must ensure only ONE wave executes this, then __syncthreads().
// ---------------------------------------------------------------------------
#if __has_builtin(__builtin_amdgcn_tensor_load_to_lds) && \
    __has_builtin(__builtin_amdgcn_s_wait_tensorcnt)
#define USE_TDM 1
#else
#define USE_TDM 0
#endif

#if USE_TDM
__device__ __forceinline__ void tdm_load_block(const void* gsrc, u32 nbytes) {
  const unsigned long long ga = (unsigned long long)(size_t)gsrc;
  const u32 u8 = nbytes >> 3;  // length in 8-byte units (data_size=3)
  u32x4 g0;
  g0.x = 1u;                                   // count=1, user descriptor
  g0.y = 0u;                                   // lds_addr = 0 (tile at base)
  g0.z = (u32)(ga & 0xffffffffu);              // global_addr[31:0]
  g0.w = (u32)((ga >> 32) & 0x01ffffffu) | (2u << 30);  // addr[56:32] | type=2
  i32x8 g1;
  g1[0] = (int)(3u << 16);                     // workgroup_mask=0, data_size=8B
  g1[1] = (int)((u8 & 0xffffu) << 16);         // tensor_dim0[15:0] @63:48
  g1[2] = (int)((u8 >> 16) & 0xffffu) |        // tensor_dim0[31:16]
          (int)(1u << 16);                     // tensor_dim1 = 1
  g1[3] = (int)(u8 << 16);                     // tensor_dim1 hi | tile_dim0=u8
  g1[4] = 1;                                   // tile_dim1 = 1, tile_dim2 = 0
  g1[5] = (int)u8;                             // tensor_dim0_stride[31:0]
  g1[6] = 0;
  g1[7] = 0;
  i32x4 z4 = {0, 0, 0, 0};
#if __has_include(<hip/amd_detail/amd_gfx1250_TDM.h>)
  i32x8 z8 = {0, 0, 0, 0, 0, 0, 0, 0};
  __builtin_amdgcn_tensor_load_to_lds(g0, g1, z4, z4, z8, 0);
#else
  __builtin_amdgcn_tensor_load_to_lds(g0, g1, z4, z4, 0);
#endif
  __builtin_amdgcn_s_wait_tensorcnt(0);
}
#endif

// Stage the packed obs tile (16KB contiguous) into LDS offset 0.
__device__ __forceinline__ void stage_obs(const _Float16* gsrc,
                                          _Float16* obs_lds, int tid) {
#if USE_TDM
  if (tid < 32) {  // one wave issues the tensor DMA (per-wave instruction)
    tdm_load_block(gsrc, (u32)(16 * D * 2));
  }
#else
  const u32* src = (const u32*)gsrc;
  u32* dst = (u32*)obs_lds;
  for (int i = tid; i < 16 * D / 2; i += 256) dst[i] = src[i];
#endif
  __syncthreads();
}

// ---------------------------------------------------------------------------
// Prep kernels
// ---------------------------------------------------------------------------
// Pack obs [B,D] fp32 row-major into per-16-row-tile A-fragment layout f16:
// dst[((bt*numKT + kt)*32 + lane)*16 + v*2 + {0,1}] — each tile is a
// contiguous 16KB block, each lane's fragment slice is 32B contiguous.
__global__ __launch_bounds__(256) void pack_obs_kernel(
    const float* __restrict__ obs, _Float16* __restrict__ dst) {
  const int numKT = D / 32;
  const int total = (BB / 16) * numKT * 32 * 8; // pair-units
  for (int idx = blockIdx.x * blockDim.x + threadIdx.x; idx < total;
       idx += gridDim.x * blockDim.x) {
    int v = idx & 7;
    int lane = (idx >> 3) & 31;
    int rem = idx >> 8;
    int kt = rem % numKT;
    int bt = rem / numKT;
    int row = bt * 16 + (lane & 15);
    int c = kt * 32 + koff_of(v, (lane >> 4) << 3);
    const float* src = obs + (size_t)row * D + c;
    dst[(size_t)idx * 2]     = (_Float16)src[0];
    dst[(size_t)idx * 2 + 1] = (_Float16)src[1];
  }
}

// Pack W [k][Dk rows(K-dim)][N cols] fp32 row-major into WMMA B-frag layout f16.
__global__ __launch_bounds__(256) void pack_w_kernel(const float* __restrict__ W,
                                                     _Float16* __restrict__ dst,
                                                     int kc, int Dk, int N) {
  const int numKT = Dk / 32;
  const int numNT = N / 16;
  const int total = kc * numNT * numKT * 32 * 8; // pair-units
  for (int idx = blockIdx.x * blockDim.x + threadIdx.x; idx < total;
       idx += gridDim.x * blockDim.x) {
    int v = idx & 7;
    int lane = (idx >> 3) & 31;
    int rem = idx >> 8;
    int kt = rem % numKT; rem /= numKT;
    int nt = rem % numNT; rem /= numNT;
    int k = rem;
    int krow = kt * 32 + koff_of(v, (lane >> 4) << 3);
    int n = nt * 16 + (lane & 15);
    const float* src = W + (size_t)k * Dk * N;
    dst[(size_t)idx * 2]     = (_Float16)src[(size_t)krow * N + n];
    dst[(size_t)idx * 2 + 1] = (_Float16)src[(size_t)(krow + 1) * N + n];
  }
}

// ---------------------------------------------------------------------------
// Pass 1: per-(k, 16-row tile): fused MLP -> logits tile in LDS -> softmax
// stats.  Emits logZ[k,b] and action_logprob[k,b].
// grid = KC * (BB/16), block = 256 (8 waves)
// dynamic LDS: packed obs (16KB) + packed h (16KB) + logits (64KB) = 96KB
// ---------------------------------------------------------------------------
__global__ __launch_bounds__(256) void stats_kernel(
    const _Float16* __restrict__ obs_pk, const _Float16* __restrict__ W1p,
    const _Float16* __restrict__ W2p, const float* __restrict__ b1,
    const float* __restrict__ b2, const int* __restrict__ actions,
    float* __restrict__ logZ, float* __restrict__ alp) {
  extern __shared__ char smem[];
  _Float16* obs_lds = (_Float16*)smem;                       // packed A-frags
  _Float16* h_pk    = (_Float16*)(smem + 16 * D * 2);        // packed A-frags
  float*    l_lds   = (float*)(smem + 16 * D * 2 + 16 * H * 2); // [16][AC]

  const int k    = blockIdx.x >> 9;     // / (BB/16)
  const int bt   = blockIdx.x & 511;
  const int row0 = bt << 4;
  const int wv   = threadIdx.x >> 5;
  const int lane = threadIdx.x & 31;
  const int col  = lane & 15;
  const int rj0  = (lane >> 4) << 3;    // row base for C/D frags

  const _Float16* W1k = W1p + (size_t)k * D * H;
  const _Float16* W2k = W2p + (size_t)k * D * AC;

  stage_obs(obs_pk + (size_t)bt * 16 * D, obs_lds, threadIdx.x);

  // ---- phase 1: h = relu(obs @ W1 + b1), 32 N-tiles, 4 per wave ----
  {
    v8f acc[4] = {};
    for (int kt = 0; kt < D / 32; ++kt) {
      v16h af = load_a_packed(obs_lds, kt, lane);
#pragma unroll
      for (int i = 0; i < 4; ++i) {
        v16h bf = load_b(W1k, D / 32, wv + (i << 3), kt, lane);
        acc[i] = wmma16(af, bf, acc[i]);
      }
    }
#pragma unroll
    for (int i = 0; i < 4; ++i) {
      const int nt = wv + (i << 3);
      const float bias = b1[k * H + nt * 16 + col];
#pragma unroll
      for (int j = 0; j < 8; ++j) {
        float x = acc[i][j] + bias;
        x = x > 0.f ? x : 0.f;
        h_pk[apack_off(rj0 + j, nt * 16 + col)] = (_Float16)x;
      }
    }
  }
  __syncthreads();

  // ---- phase 2: logits = h @ W2 + b2, 64 N-tiles, 8 per wave ----
  {
    v8f acc[8] = {};
    for (int kt = 0; kt < H / 32; ++kt) {
      v16h af = load_a_packed(h_pk, kt, lane);
#pragma unroll
      for (int i = 0; i < 8; ++i) {
        v16h bf = load_b(W2k, H / 32, wv + (i << 3), kt, lane);
        acc[i] = wmma16(af, bf, acc[i]);
      }
    }
#pragma unroll
    for (int i = 0; i < 8; ++i) {
      const int nt = wv + (i << 3);
      const float bias = b2[k * AC + nt * 16 + col];
#pragma unroll
      for (int j = 0; j < 8; ++j)
        l_lds[(size_t)(rj0 + j) * AC + nt * 16 + col] = acc[i][j] + bias;
    }
  }
  __syncthreads();

  // ---- phase 3: per-row logsumexp + gather action logit ----
  for (int r = wv; r < 16; r += 8) {
    const float* rowp = l_lds + (size_t)r * AC;
    float m = -3.0e38f;
    for (int c = lane; c < AC; c += 32) m = fmaxf(m, rowp[c]);
#pragma unroll
    for (int off = 16; off >= 1; off >>= 1) m = fmaxf(m, __shfl_xor(m, off, 32));
    float s = 0.f;
    for (int c = lane; c < AC; c += 32) s += __expf(rowp[c] - m);
#pragma unroll
    for (int off = 16; off >= 1; off >>= 1) s += __shfl_xor(s, off, 32);
    if (lane == 0) {
      const int b = row0 + r;
      const float lz = m + __logf(s);
      logZ[k * BB + b] = lz;
      alp[k * BB + b] = rowp[actions[b]] - lz;
    }
  }
}

// ---------------------------------------------------------------------------
// Pass 2: sequential mixture scan per sequence. grid = SQ, block = 32.
// lanes 0..7 carry components.
// ---------------------------------------------------------------------------
__global__ __launch_bounds__(32) void mix_kernel(
    const float* __restrict__ start, const float* __restrict__ alp,
    float* __restrict__ mix, float* __restrict__ fin_out) {
  const int s = blockIdx.x;
  const int lane = threadIdx.x;
  const bool act = lane < KC;
  const float st = act ? start[s * KC + lane] : -3.0e38f;
  float cum = 0.f;
  for (int t = 0; t < TT; ++t) {
    const int b = s * TT + t;
    const float ml = st + cum;         // uses cum BEFORE this step (shift)
    float m = ml;
#pragma unroll
    for (int off = 4; off >= 1; off >>= 1) m = fmaxf(m, __shfl_xor(m, off, 8));
    float ss = __expf(ml - m);
#pragma unroll
    for (int off = 4; off >= 1; off >>= 1) ss += __shfl_xor(ss, off, 8);
    const float lse = m + __logf(ss);
    if (act) mix[b * KC + lane] = ml - lse;
    cum += act ? alp[lane * BB + b] : 0.f;
  }
  const float fl = st + cum;
  float m = fl;
#pragma unroll
  for (int off = 4; off >= 1; off >>= 1) m = fmaxf(m, __shfl_xor(m, off, 8));
  float ss = __expf(fl - m);
#pragma unroll
  for (int off = 4; off >= 1; off >>= 1) ss += __shfl_xor(ss, off, 8);
  if (act) fin_out[s * KC + lane] = fl - (m + __logf(ss));
}

// ---------------------------------------------------------------------------
// Pass 3: recompute per-tile logits for all K, online logsumexp-over-K into
// model_out.  grid = BB/16, block = 256 (8 waves).
// ---------------------------------------------------------------------------
__global__ __launch_bounds__(256) void out_kernel(
    const _Float16* __restrict__ obs_pk, const _Float16* __restrict__ W1p,
    const _Float16* __restrict__ W2p, const float* __restrict__ b1,
    const float* __restrict__ b2, const float* __restrict__ logZ,
    const float* __restrict__ mix, float* __restrict__ out) {
  __shared__ _Float16 obs_lds[16 * D];  // must stay first (TDM lds_addr = 0)
  __shared__ _Float16 h_pk[16 * H];     // packed A-frags

  const int bt   = blockIdx.x;
  const int row0 = bt << 4;
  const int wv   = threadIdx.x >> 5;
  const int lane = threadIdx.x & 31;
  const int col  = lane & 15;
  const int rj0  = (lane >> 4) << 3;

  stage_obs(obs_pk + (size_t)bt * 16 * D, obs_lds, threadIdx.x);

  float m8[8][8], s8[8][8];
#pragma unroll
  for (int i = 0; i < 8; ++i)
#pragma unroll
    for (int j = 0; j < 8; ++j) { m8[i][j] = -3.0e38f; s8[i][j] = 0.f; }

  for (int k = 0; k < KC; ++k) {
    const _Float16* W1k = W1p + (size_t)k * D * H;
    const _Float16* W2k = W2p + (size_t)k * D * AC;

    // h tile
    v8f acc1[4] = {};
    for (int kt = 0; kt < D / 32; ++kt) {
      v16h af = load_a_packed(obs_lds, kt, lane);
#pragma unroll
      for (int i = 0; i < 4; ++i) {
        v16h bf = load_b(W1k, D / 32, wv + (i << 3), kt, lane);
        acc1[i] = wmma16(af, bf, acc1[i]);
      }
    }
#pragma unroll
    for (int i = 0; i < 4; ++i) {
      const int nt = wv + (i << 3);
      const float bias = b1[k * H + nt * 16 + col];
#pragma unroll
      for (int j = 0; j < 8; ++j) {
        float x = acc1[i][j] + bias;
        x = x > 0.f ? x : 0.f;
        h_pk[apack_off(rj0 + j, nt * 16 + col)] = (_Float16)x;
      }
    }
    __syncthreads();

    // logits tile + online logsumexp over k
    v8f acc2[8] = {};
    for (int kt = 0; kt < H / 32; ++kt) {
      v16h af = load_a_packed(h_pk, kt, lane);
#pragma unroll
      for (int i = 0; i < 8; ++i) {
        v16h bf = load_b(W2k, H / 32, wv + (i << 3), kt, lane);
        acc2[i] = wmma16(af, bf, acc2[i]);
      }
    }
    float c8[8];
#pragma unroll
    for (int j = 0; j < 8; ++j) {
      const int b = row0 + rj0 + j;
      c8[j] = mix[b * KC + k] - logZ[k * BB + b];
    }
#pragma unroll
    for (int i = 0; i < 8; ++i) {
      const int nt = wv + (i << 3);
      const float bias = b2[k * AC + nt * 16 + col];
#pragma unroll
      for (int j = 0; j < 8; ++j) {
        const float x = acc2[i][j] + bias + c8[j];
        const float mo = m8[i][j];
        const float mn = fmaxf(mo, x);
        s8[i][j] = s8[i][j] * __expf(mo - mn) + __expf(x - mn);
        m8[i][j] = mn;
      }
    }
    __syncthreads();  // protect h_pk before next k overwrites it
  }

#pragma unroll
  for (int i = 0; i < 8; ++i) {
    const int nt = wv + (i << 3);
#pragma unroll
    for (int j = 0; j < 8; ++j) {
      const int b = row0 + rj0 + j;
      out[(size_t)b * AC + nt * 16 + col] = m8[i][j] + __logf(s8[i][j]);
    }
  }
}

// ---------------------------------------------------------------------------
// Host launcher
// ---------------------------------------------------------------------------
extern "C" void kernel_launch(void* const* d_in, const int* in_sizes, int n_in,
                              void* d_out, int out_size, void* d_ws,
                              size_t ws_size, hipStream_t stream) {
  const float* obs     = (const float*)d_in[0];   // [B,D]
  const int*   actions = (const int*)d_in[1];     // [B]
  const float* start   = (const float*)d_in[2];   // [S,K]
  const float* W1      = (const float*)d_in[3];   // [K,D,H]
  const float* b1      = (const float*)d_in[4];   // [K,H]
  const float* W2      = (const float*)d_in[5];   // [K,H,A]
  const float* b2      = (const float*)d_in[6];   // [K,A]
  float* out = (float*)d_out;                     // [B,A] ++ [S,K]

  // workspace carve-up (all 256B aligned by construction)
  char* ws = (char*)d_ws;
  _Float16* obs_pk = (_Float16*)ws;                      ws += (size_t)BB * D * 2;
  _Float16* W1p    = (_Float16*)ws;                      ws += (size_t)KC * D * H * 2;
  _Float16* W2p    = (_Float16*)ws;                      ws += (size_t)KC * D * AC * 2;
  float* logZ = (float*)ws;                              ws += (size_t)KC * BB * 4;
  float* alp  = (float*)ws;                              ws += (size_t)KC * BB * 4;
  float* mixb = (float*)ws;                              ws += (size_t)BB * KC * 4;

  // 1) pack obs into per-tile A-fragment layout (f16)
  pack_obs_kernel<<<2048, 256, 0, stream>>>(obs, obs_pk);
  // 2) pack weights into WMMA B-fragment layout
  pack_w_kernel<<<2048, 256, 0, stream>>>(W1, W1p, KC, D, H);
  pack_w_kernel<<<4096, 256, 0, stream>>>(W2, W2p, KC, H, AC);
  // 3) per-(k,row-tile) fused MLP + softmax stats
  const size_t smem_stats = (size_t)16 * D * 2 + 16 * H * 2 + 16 * AC * 4; // 96KB
  stats_kernel<<<KC * (BB / 16), 256, smem_stats, stream>>>(
      obs_pk, W1p, W2p, b1, b2, actions, logZ, alp);
  // 4) sequential mixture posterior scan (+ final mixture logprobs tail)
  mix_kernel<<<SQ, 32, 0, stream>>>(start, alp, mixb, out + (size_t)BB * AC);
  // 5) recompute + combine into model_out
  out_kernel<<<BB / 16, 256, 0, stream>>>(obs_pk, W1p, W2p, b1, b2, logZ, mixb,
                                          out);
}
